// PointNetPP_SSG_Seg_30056181138031
// MI455X (gfx1250) — matrix operand order repaired and checked by reading, not compiled, not run on hardware
//
#include <hip/hip_runtime.h>
#include <hip/hip_bf16.h>

// ---------------------------------------------------------------------------
// PointNet++ SSG segmentation for MI455X (gfx1250, wave32, WMMA).
// All MLP layers -> fused f16 WMMA GEMM (v_wmma_f32_16x16x32_f16) + BN + ReLU.
// GEMM waves own 32x64 tiles (8 WMMAs / K-chunk), tile counts are template
// constants so the K-loop is branch-free (no accumulator copy storms).
// ---------------------------------------------------------------------------

typedef _Float16 h16;
typedef __attribute__((ext_vector_type(16))) _Float16 v16h;
typedef __attribute__((ext_vector_type(8)))  _Float16 v8h;
typedef __attribute__((ext_vector_type(8)))  float    v8f;

// ------------------------- weight / BN preparation -------------------------
__global__ void prep_layer_kernel(const float* __restrict__ W,
                                  const float* __restrict__ gamma,
                                  const float* __restrict__ beta,
                                  const float* __restrict__ mean,
                                  const float* __restrict__ var,
                                  h16* __restrict__ wf,
                                  float* __restrict__ sc,
                                  float* __restrict__ bi,
                                  int cin, int cout, int Kpad, int Npad) {
  int t = blockIdx.x * blockDim.x + threadIdx.x;
  if (t >= Npad * Kpad) return;
  int n = t / Kpad, k = t % Kpad;
  float w = (n < cout && k < cin) ? W[n * cin + k] : 0.0f;
  wf[t] = (h16)w;
  if (k == 0) {
    if (n < cout && gamma) {
      float s = gamma[n] * rsqrtf(var[n] + 1e-5f);
      sc[n] = s;
      bi[n] = beta[n] - mean[n] * s;
    } else {
      sc[n] = 0.0f;
      bi[n] = 0.0f;
    }
  }
}

// ----------------------------- FPS (per batch) -----------------------------
#define FPS_BLOCK 256
__global__ __launch_bounds__(FPS_BLOCK) void fps_kernel(
    const float* __restrict__ xyz, int N, int npoint,
    int* __restrict__ out_idx, float* __restrict__ new_xyz) {
  __shared__ float sdist[4096];
  __shared__ float wmaxv[FPS_BLOCK / 32];
  __shared__ int   wmaxi[FPS_BLOCK / 32];
  __shared__ int   sfar;
  const int b   = blockIdx.x;
  const int tid = threadIdx.x;
  const float* p = xyz + (size_t)b * N * 3;

  for (int i = tid; i < N; i += FPS_BLOCK) sdist[i] = 1e10f;
  if (tid == 0) sfar = 0;
  __syncthreads();

  for (int s = 0; s < npoint; ++s) {
    const int far = sfar;
    const float cx = p[far * 3 + 0];
    const float cy = p[far * 3 + 1];
    const float cz = p[far * 3 + 2];
    if (tid == 0) {
      out_idx[(size_t)b * npoint + s] = far;
      float* q = new_xyz + ((size_t)b * npoint + s) * 3;
      q[0] = cx; q[1] = cy; q[2] = cz;
    }
    float best = -1.0f;
    int   bi   = 0;
    for (int i = tid; i < N; i += FPS_BLOCK) {
      float dx = p[i * 3 + 0] - cx;
      float dy = p[i * 3 + 1] - cy;
      float dz = p[i * 3 + 2] - cz;
      float d  = dx * dx + dy * dy + dz * dz;
      float dm = fminf(sdist[i], d);
      sdist[i] = dm;
      if (dm > best) { best = dm; bi = i; }
    }
    #pragma unroll
    for (int off = 16; off > 0; off >>= 1) {
      float ov = __shfl_down(best, off, 32);
      int   oi = __shfl_down(bi,   off, 32);
      if (ov > best || (ov == best && oi < bi)) { best = ov; bi = oi; }
    }
    if ((tid & 31) == 0) { wmaxv[tid >> 5] = best; wmaxi[tid >> 5] = bi; }
    __syncthreads();
    if (tid == 0) {
      float bv = wmaxv[0]; int bx = wmaxi[0];
      for (int w = 1; w < FPS_BLOCK / 32; ++w)
        if (wmaxv[w] > bv || (wmaxv[w] == bv && wmaxi[w] < bx)) { bv = wmaxv[w]; bx = wmaxi[w]; }
      sfar = bx;
    }
    __syncthreads();
  }
}

// ------------------------------- ball query --------------------------------
__global__ void ball_query_kernel(const float* __restrict__ xyz,
                                  const float* __restrict__ new_xyz,
                                  int* __restrict__ idx,
                                  int B, int N, int S, float r2, int nsample) {
  int t = blockIdx.x * blockDim.x + threadIdx.x;
  if (t >= B * S) return;
  int b = t / S;
  const float* q = new_xyz + (size_t)t * 3;
  const float* p = xyz + (size_t)b * N * 3;
  int* o = idx + (size_t)t * nsample;
  int cnt = 0, first = 0;
  for (int i = 0; i < N && cnt < nsample; ++i) {
    float dx = p[3 * i + 0] - q[0];
    float dy = p[3 * i + 1] - q[1];
    float dz = p[3 * i + 2] - q[2];
    float d = dx * dx + dy * dy + dz * dz;
    if (d <= r2) {
      if (cnt == 0) first = i;
      o[cnt++] = i;
    }
  }
  for (; cnt < nsample; ++cnt) o[cnt] = first;
}

// ------------------------- grouping + f16 concat ---------------------------
__global__ void group_kernel(const float* __restrict__ xyz,
                             const float* __restrict__ new_xyz,
                             const int* __restrict__ idx,
                             const h16* __restrict__ feats, int C,
                             h16* __restrict__ act, int Kpad,
                             int B, int N, int S, int ns) {
  long long t = (long long)blockIdx.x * blockDim.x + threadIdx.x;
  long long total = (long long)B * S * ns * Kpad;
  if (t >= total) return;
  int k = (int)(t % Kpad);
  long long r = t / Kpad;
  int j  = (int)(r % ns);
  long long bs = r / ns;
  int b = (int)(bs / S);
  int pi = idx[bs * ns + j];
  float val = 0.0f;
  if (k < 3) {
    val = xyz[((size_t)b * N + pi) * 3 + k] - new_xyz[bs * 3 + k];
  } else if (feats && k < 3 + C) {
    val = (float)feats[((size_t)b * N + pi) * C + (k - 3)];
  }
  act[(size_t)r * Kpad + k] = (h16)val;
}

// ------------------------------- max pool ----------------------------------
__global__ void maxpool_kernel(const h16* __restrict__ act, h16* __restrict__ out,
                               int BS, int C, int ldi, int ns) {
  int t = blockIdx.x * blockDim.x + threadIdx.x;
  if (t >= BS * C) return;
  int bs = t / C, c = t % C;
  const h16* p = act + (size_t)bs * ns * ldi + c;
  float m = -1e30f;
  for (int j = 0; j < ns; ++j) m = fmaxf(m, (float)p[(size_t)j * ldi]);
  out[(size_t)bs * C + c] = (h16)m;
}

// ------------------------------ kNN (k = 3) --------------------------------
__global__ void knn3_kernel(const float* __restrict__ xyz1,
                            const float* __restrict__ xyz2,
                            int B, int S1, int S2,
                            int* __restrict__ oidx, float* __restrict__ ow) {
  int t = blockIdx.x * blockDim.x + threadIdx.x;
  if (t >= B * S1) return;
  int b = t / S1;
  const float* q = xyz1 + (size_t)t * 3;
  const float* p = xyz2 + (size_t)b * S2 * 3;
  float d0 = 1e30f, d1 = 1e30f, d2 = 1e30f;
  int   i0 = 0,     i1 = 0,     i2 = 0;
  for (int i = 0; i < S2; ++i) {
    float dx = p[3 * i + 0] - q[0];
    float dy = p[3 * i + 1] - q[1];
    float dz = p[3 * i + 2] - q[2];
    float d = dx * dx + dy * dy + dz * dz;
    if (d < d0)      { d2 = d1; i2 = i1; d1 = d0; i1 = i0; d0 = d; i0 = i; }
    else if (d < d1) { d2 = d1; i2 = i1; d1 = d;  i1 = i; }
    else if (d < d2) { d2 = d;  i2 = i; }
  }
  float w0 = 1.0f / fmaxf(d0, 1e-10f);
  float w1 = 1.0f / fmaxf(d1, 1e-10f);
  float w2 = 1.0f / fmaxf(d2, 1e-10f);
  float inv = 1.0f / (w0 + w1 + w2);
  oidx[t * 3 + 0] = i0; oidx[t * 3 + 1] = i1; oidx[t * 3 + 2] = i2;
  ow[t * 3 + 0] = w0 * inv; ow[t * 3 + 1] = w1 * inv; ow[t * 3 + 2] = w2 * inv;
}

// ------------------- feature-prop concat + interpolation -------------------
__global__ void fp_concat_kernel(const h16* __restrict__ pts1, int C1,
                                 const h16* __restrict__ pts2, int C2,
                                 const int* __restrict__ kidx,
                                 const float* __restrict__ kw,
                                 h16* __restrict__ act, int Kpad,
                                 int B, int S1, int S2) {
  long long t = (long long)blockIdx.x * blockDim.x + threadIdx.x;
  long long total = (long long)B * S1 * Kpad;
  if (t >= total) return;
  int k = (int)(t % Kpad);
  long long r = t / Kpad;
  int b = (int)(r / S1);
  float val = 0.0f;
  if (pts1 && k < C1) {
    val = (float)pts1[(size_t)r * C1 + k];
  } else if (k < C1 + C2) {
    int c = k - C1;
    const int*   ii = kidx + r * 3;
    const float* ww = kw   + r * 3;
    const h16* base = pts2 + (size_t)b * S2 * C2;
    val = ww[0] * (float)base[(size_t)ii[0] * C2 + c]
        + ww[1] * (float)base[(size_t)ii[1] * C2 + c]
        + ww[2] * (float)base[(size_t)ii[2] * C2 + c];
  }
  act[(size_t)r * Kpad + k] = (h16)val;
}

// ------------------- fused WMMA GEMM + BN + ReLU epilogue ------------------
// C[M x Npad] = A[M x Kpad](f16) * W[Npad x Kpad]^T (f16), f32 accumulate.
// Wave tile: 32(M) x NT*16(N). NT is a compile-time constant so the K-loop
// carries no branches: per K-chunk = 2 A-frag loads + NT B-frag loads +
// 2*NT back-to-back v_wmma_f32_16x16x32_f16 (B reused across both M tiles).
// A frags use contiguous K (not ISA table order): lanes 0-15 hold K 0-15,
// lanes 16-31 hold K 16-31; B: half-waves hold 16 contiguous K.
// No explicit prefetch: 4 waves/block x thousands of blocks hide latency,
// and streamed activations hit the 192MB L2 anyway.
template <int NT, bool F32OUT>
__global__ __launch_bounds__(128) void wmma_gemm_tpl(
    const h16* __restrict__ A, const h16* __restrict__ W,
    const float* __restrict__ scale, const float* __restrict__ bias,
    h16* __restrict__ outH, float* __restrict__ outF,
    int M, int Kpad, int ldo, int nvalid, int relu) {
  const int lane = threadIdx.x & 31;
  const int wv   = threadIdx.x >> 5;
  const int m0 = (blockIdx.x * 4 + wv) * 32;   // 32 M rows per wave
  const int n0 = blockIdx.y * (NT * 16);
  if (m0 >= M) return;                          // wave-uniform

  const int rsel   = lane & 15;
  const int half16 = lane >> 4;                 // 0: lanes 0-15, 1: 16-31
  const h16* arow0 = A + (size_t)(m0 + rsel) * Kpad + half16 * 16;
  const h16* arow1 = arow0 + (size_t)16 * Kpad;
  const h16* wbase = W + (size_t)(n0 + rsel) * Kpad + half16 * 16;

  v8f acc0[NT] = {};
  v8f acc1[NT] = {};

  for (int k0 = 0; k0 < Kpad; k0 += 32) {
    v8h a0lo = *(const v8h*)(arow0 + k0);
    v8h a0hi = *(const v8h*)(arow0 + k0 + 8);
    v16h a0 = __builtin_shufflevector(a0lo, a0hi, 0, 1, 2, 3, 4, 5, 6, 7,
                                                  8, 9, 10, 11, 12, 13, 14, 15);
    v8h a1lo = *(const v8h*)(arow1 + k0);
    v8h a1hi = *(const v8h*)(arow1 + k0 + 8);
    v16h a1 = __builtin_shufflevector(a1lo, a1hi, 0, 1, 2, 3, 4, 5, 6, 7,
                                                  8, 9, 10, 11, 12, 13, 14, 15);
    #pragma unroll
    for (int t = 0; t < NT; ++t) {
      v16h bfrag = *(const v16h*)(wbase + (size_t)t * 16 * Kpad + k0);
      acc0[t] = __builtin_amdgcn_wmma_f32_16x16x32_f16(
          false, a0, false, bfrag, (short)0, acc0[t], false, false);
      acc1[t] = __builtin_amdgcn_wmma_f32_16x16x32_f16(
          false, a1, false, bfrag, (short)0, acc1[t], false, false);
    }
  }

  // C/D layout: lane -> N = lane&15, M = vgpr + (lane>>4)*8
  const int mr0 = m0 + half16 * 8;
  #pragma unroll
  for (int t = 0; t < NT; ++t) {
    int n = n0 + t * 16 + rsel;
    float s  = (scale && n < nvalid) ? scale[n] : 1.0f;
    float bb = (bias  && n < nvalid) ? bias[n]  : 0.0f;
    #pragma unroll
    for (int v = 0; v < 8; ++v) {
      float x0 = acc0[t][v] * s + bb;
      float x1 = acc1[t][v] * s + bb;
      if (relu) { x0 = fmaxf(x0, 0.0f); x1 = fmaxf(x1, 0.0f); }
      if constexpr (F32OUT) {
        if (n < nvalid) {
          outF[(size_t)(mr0 + v) * ldo + n]      = x0;
          outF[(size_t)(mr0 + 16 + v) * ldo + n] = x1;
        }
      } else {
        outH[(size_t)(mr0 + v) * ldo + n]      = (h16)x0;
        outH[(size_t)(mr0 + 16 + v) * ldo + n] = (h16)x1;
      }
    }
  }
}

// ---------------------------------------------------------------------------
// Host orchestration
// ---------------------------------------------------------------------------
namespace {
struct LayerRT {
  const float *W, *g, *bet, *mu, *var;
  int cin, cout, Kpad, Npad;
  h16* wf; float* sc; float* bi;
};
// d_in leaf order (setup_inputs insertion order):
// 0:xyz; sa1 1-15; sa2 16-30; sa3 31-45; fp3 46-55; fp2 56-65; fp1 66-80;
// head: W1=81, bn1{W=82,g=83,b=84,m=85,v=86}, W2=87, b2=88.
static const int SPEC[18][7] = {
  { 1,  2,  3,  4,  5,    3,   64}, { 6,  7,  8,  9, 10,   64,   64},
  {11, 12, 13, 14, 15,   64,  128},
  {16, 17, 18, 19, 20,  131,  128}, {21, 22, 23, 24, 25,  128,  128},
  {26, 27, 28, 29, 30,  128,  256},
  {31, 32, 33, 34, 35,  259,  256}, {36, 37, 38, 39, 40,  256,  512},
  {41, 42, 43, 44, 45,  512, 1024},
  {46, 47, 48, 49, 50, 1280,  256}, {51, 52, 53, 54, 55,  256,  256},
  {56, 57, 58, 59, 60,  384,  256}, {61, 62, 63, 64, 65,  256,  128},
  {66, 67, 68, 69, 70,  128,  128}, {71, 72, 73, 74, 75,  128,  128},
  {76, 77, 78, 79, 80,  128,  128},
  {81, 83, 84, 85, 86,  128,  128},      // head: W1 + bn1 params
  {87, -1, -1, -1, -1,  128,   13},      // head: W2 (plain bias b2)
};
} // namespace

extern "C" void kernel_launch(void* const* d_in, const int* in_sizes, int n_in,
                              void* d_out, int out_size, void* d_ws, size_t ws_size,
                              hipStream_t stream) {
  (void)in_sizes; (void)out_size; (void)ws_size;
  if (n_in < 89) return;

  const int B = 8, N0 = 4096, S1 = 1024, S2 = 256, S3 = 64, NS = 32;
  const float* xyz = (const float*)d_in[0];
  float* out = (float*)d_out;

  size_t off = 0;
  auto alloc = [&](size_t bytes) -> void* {
    off = (off + 255) & ~(size_t)255;
    void* p = (char*)d_ws + off;
    off += bytes;
    return p;
  };

  // ---- layer tables + weight prep (weights tiny -> L2 resident) ----
  LayerRT L[18];
  for (int i = 0; i < 18; ++i) {
    const int* s = SPEC[i];
    L[i].W   = (const float*)d_in[s[0]];
    L[i].g   = s[1] >= 0 ? (const float*)d_in[s[1]] : nullptr;
    L[i].bet = s[2] >= 0 ? (const float*)d_in[s[2]] : nullptr;
    L[i].mu  = s[3] >= 0 ? (const float*)d_in[s[3]] : nullptr;
    L[i].var = s[4] >= 0 ? (const float*)d_in[s[4]] : nullptr;
    L[i].cin = s[5]; L[i].cout = s[6];
    L[i].Kpad = (s[5] + 31) & ~31;
    L[i].Npad = (s[6] + 15) & ~15;
    L[i].wf = (h16*)alloc((size_t)L[i].Kpad * L[i].Npad * sizeof(h16));
    L[i].sc = (float*)alloc((size_t)L[i].Npad * sizeof(float));
    L[i].bi = (float*)alloc((size_t)L[i].Npad * sizeof(float));
    int tot = L[i].Kpad * L[i].Npad;
    prep_layer_kernel<<<(tot + 255) / 256, 256, 0, stream>>>(
        L[i].W, L[i].g, L[i].bet, L[i].mu, L[i].var,
        L[i].wf, L[i].sc, L[i].bi, L[i].cin, L[i].cout, L[i].Kpad, L[i].Npad);
  }

  // ---- scratch buffers ----
  h16* actA = (h16*)alloc((size_t)262144 * 64 * sizeof(h16));   // 32 MB
  h16* actB = (h16*)alloc((size_t)262144 * 128 * sizeof(h16));  // 64 MB
  float* nx1 = (float*)alloc((size_t)B * S1 * 3 * sizeof(float));
  float* nx2 = (float*)alloc((size_t)B * S2 * 3 * sizeof(float));
  float* nx3 = (float*)alloc((size_t)B * S3 * 3 * sizeof(float));
  int* fidx1 = (int*)alloc((size_t)B * S1 * sizeof(int));
  int* fidx2 = (int*)alloc((size_t)B * S2 * sizeof(int));
  int* fidx3 = (int*)alloc((size_t)B * S3 * sizeof(int));
  int* bidx1 = (int*)alloc((size_t)B * S1 * NS * sizeof(int));
  int* bidx2 = (int*)alloc((size_t)B * S2 * NS * sizeof(int));
  int* bidx3 = (int*)alloc((size_t)B * S3 * NS * sizeof(int));
  h16* l1_pts = (h16*)alloc((size_t)B * S1 * 128 * sizeof(h16));
  h16* l2_pts = (h16*)alloc((size_t)B * S2 * 256 * sizeof(h16));
  h16* l3_pts = (h16*)alloc((size_t)B * S3 * 1024 * sizeof(h16));
  h16* l2_new = (h16*)alloc((size_t)B * S2 * 256 * sizeof(h16));
  h16* l1_new = (h16*)alloc((size_t)B * S1 * 128 * sizeof(h16));
  h16* l0_new = (h16*)alloc((size_t)B * N0 * 128 * sizeof(h16));
  int*   ki3 = (int*)alloc((size_t)B * S2 * 3 * sizeof(int));
  float* kw3 = (float*)alloc((size_t)B * S2 * 3 * sizeof(float));
  int*   ki2 = (int*)alloc((size_t)B * S1 * 3 * sizeof(int));
  float* kw2 = (float*)alloc((size_t)B * S1 * 3 * sizeof(float));
  int*   ki1 = (int*)alloc((size_t)B * N0 * 3 * sizeof(int));
  float* kw1 = (float*)alloc((size_t)B * N0 * 3 * sizeof(float));

  // grid: 128 M rows / block (4 waves x 32), N in NT*16 columns per block
  auto gemm = [&](const h16* Ain, int li, h16* oH, float* oF,
                  int M, int ldo, int nvalid, int relu, const float* bias_ovr) {
    const float* sc = bias_ovr ? nullptr : L[li].sc;
    const float* bi = bias_ovr ? bias_ovr : L[li].bi;
    if (oF) {
      dim3 grid(M / 128, (L[li].Npad + 15) / 16);
      wmma_gemm_tpl<1, true><<<grid, 128, 0, stream>>>(
          Ain, L[li].wf, sc, bi, nullptr, oF, M, L[li].Kpad, ldo, nvalid, relu);
    } else {
      dim3 grid(M / 128, L[li].Npad / 64);
      wmma_gemm_tpl<4, false><<<grid, 128, 0, stream>>>(
          Ain, L[li].wf, sc, bi, oH, nullptr, M, L[li].Kpad, ldo, nvalid, relu);
    }
  };
  auto grp = [&](const float* px, const float* nxz, const int* idx,
                 const h16* feats, int C, h16* act, int Kpad, int N, int S) {
    long long tot = (long long)B * S * NS * Kpad;
    group_kernel<<<(unsigned)((tot + 255) / 256), 256, 0, stream>>>(
        px, nxz, idx, feats, C, act, Kpad, B, N, S, NS);
  };

  // =================== SA1: 4096 -> 1024, r=0.1, [3,64,64,128] =============
  fps_kernel<<<B, FPS_BLOCK, 0, stream>>>(xyz, N0, S1, fidx1, nx1);
  ball_query_kernel<<<(B * S1 + 255) / 256, 256, 0, stream>>>(
      xyz, nx1, bidx1, B, N0, S1, 0.01f, NS);
  grp(xyz, nx1, bidx1, nullptr, 0, actA, 32, N0, S1);
  const int M1 = B * S1 * NS; // 262144
  gemm(actA, 0, actB, nullptr, M1, 64, 64, 1, nullptr);
  gemm(actB, 1, actA, nullptr, M1, 64, 64, 1, nullptr);
  gemm(actA, 2, actB, nullptr, M1, 128, 128, 1, nullptr);
  maxpool_kernel<<<(B * S1 * 128 + 255) / 256, 256, 0, stream>>>(
      actB, l1_pts, B * S1, 128, 128, NS);

  // =================== SA2: 1024 -> 256, r=0.2, [131,128,128,256] ==========
  fps_kernel<<<B, FPS_BLOCK, 0, stream>>>(nx1, S1, S2, fidx2, nx2);
  ball_query_kernel<<<(B * S2 + 255) / 256, 256, 0, stream>>>(
      nx1, nx2, bidx2, B, S1, S2, 0.04f, NS);
  grp(nx1, nx2, bidx2, l1_pts, 128, actA, 160, S1, S2);
  const int M2 = B * S2 * NS; // 65536
  gemm(actA, 3, actB, nullptr, M2, 128, 128, 1, nullptr);
  gemm(actB, 4, actA, nullptr, M2, 128, 128, 1, nullptr);
  gemm(actA, 5, actB, nullptr, M2, 256, 256, 1, nullptr);
  maxpool_kernel<<<(B * S2 * 256 + 255) / 256, 256, 0, stream>>>(
      actB, l2_pts, B * S2, 256, 256, NS);

  // =================== SA3: 256 -> 64, r=0.4, [259,256,512,1024] ===========
  fps_kernel<<<B, FPS_BLOCK, 0, stream>>>(nx2, S2, S3, fidx3, nx3);
  ball_query_kernel<<<(B * S3 + 255) / 256, 256, 0, stream>>>(
      nx2, nx3, bidx3, B, S2, S3, 0.16f, NS);
  grp(nx2, nx3, bidx3, l2_pts, 256, actA, 288, S2, S3);
  const int M3 = B * S3 * NS; // 16384
  gemm(actA, 6, actB, nullptr, M3, 256, 256, 1, nullptr);
  gemm(actB, 7, actA, nullptr, M3, 512, 512, 1, nullptr);
  gemm(actA, 8, actB, nullptr, M3, 1024, 1024, 1, nullptr);
  maxpool_kernel<<<(B * S3 * 1024 + 255) / 256, 256, 0, stream>>>(
      actB, l3_pts, B * S3, 1024, 1024, NS);

  // =================== FP3: (l2, l3) -> l2_new [1280,256,256] ==============
  knn3_kernel<<<(B * S2 + 255) / 256, 256, 0, stream>>>(nx2, nx3, B, S2, S3, ki3, kw3);
  {
    long long tot = (long long)B * S2 * 1280;
    fp_concat_kernel<<<(unsigned)((tot + 255) / 256), 256, 0, stream>>>(
        l2_pts, 256, l3_pts, 1024, ki3, kw3, actA, 1280, B, S2, S3);
  }
  gemm(actA, 9, actB, nullptr, B * S2, 256, 256, 1, nullptr);
  gemm(actB, 10, l2_new, nullptr, B * S2, 256, 256, 1, nullptr);

  // =================== FP2: (l1, l2_new) -> l1_new [384,256,128] ===========
  knn3_kernel<<<(B * S1 + 255) / 256, 256, 0, stream>>>(nx1, nx2, B, S1, S2, ki2, kw2);
  {
    long long tot = (long long)B * S1 * 384;
    fp_concat_kernel<<<(unsigned)((tot + 255) / 256), 256, 0, stream>>>(
        l1_pts, 128, l2_new, 256, ki2, kw2, actA, 384, B, S1, S2);
  }
  gemm(actA, 11, actB, nullptr, B * S1, 256, 256, 1, nullptr);
  gemm(actB, 12, l1_new, nullptr, B * S1, 128, 128, 1, nullptr);

  // =================== FP1: (xyz, l1_new) -> l0_new [128,128,128] ==========
  knn3_kernel<<<(B * N0 + 255) / 256, 256, 0, stream>>>(xyz, nx1, B, N0, S1, ki1, kw1);
  {
    long long tot = (long long)B * N0 * 128;
    fp_concat_kernel<<<(unsigned)((tot + 255) / 256), 256, 0, stream>>>(
        nullptr, 0, l1_new, 128, ki1, kw1, actA, 128, B, N0, S1);
  }
  const int M0 = B * N0; // 32768
  gemm(actA, 13, actB, nullptr, M0, 128, 128, 1, nullptr);
  gemm(actB, 14, actA, nullptr, M0, 128, 128, 1, nullptr);
  gemm(actA, 15, l0_new, nullptr, M0, 128, 128, 1, nullptr);

  // =================== head: 128 -> 128(bn,relu) -> 13 (+bias) =============
  gemm(l0_new, 16, actA, nullptr, M0, 128, 128, 1, nullptr);
  gemm(actA, 17, nullptr, out, M0, /*ldo=*/13, /*nvalid=*/13, /*relu=*/0,
       /*bias override=*/(const float*)d_in[88]);
}
